// MHA_38354057953519
// MI455X (gfx1250) — matrix operand order repaired and checked by reading, not compiled, not run on hardware
//
#include <hip/hip_runtime.h>

// ---------------------------------------------------------------------------
// MHA forward for gfx1250 (CDNA5, wave32, WMMA bf16 16x16x32, f32 accum)
//   h      : [2048, 2, 1024] f32
//   w_qkv  : [3072, 1024]    f32
//   out    : attn_vec [2048,2,1024] f32 ++ attn_prob [2048,2048,2,16] f32
// Software-pipelined with immediate-offset addressing (no pointer arrays ->
// no scratch spills in the hot loops).
// ---------------------------------------------------------------------------

#define QLEN   2048
#define BSZ    2
#define NHEAD  16
#define DHEAD  64
#define DMODEL 1024
#define NQKV   3072           // 3 * NHEAD * DHEAD
#define MROWS  (QLEN * BSZ)   // 4096
#define SCALE_Q 0.125f        // 1/sqrt(64)

typedef __attribute__((ext_vector_type(16))) __bf16 v16bf;
typedef __attribute__((ext_vector_type(8)))  float  v8f;

union BFrag {            // 32 bytes: one bf16 WMMA A/B operand (8 VGPRs)
    v16bf v;
    uint4 q[2];
    unsigned short u[16];
};

__device__ __forceinline__ unsigned short f2bf(float f) {
    unsigned int x = __float_as_uint(f);
    x += 0x7FFFu + ((x >> 16) & 1u);     // round-to-nearest-even
    return (unsigned short)(x >> 16);
}

__device__ __forceinline__ v8f wmma_bf16(const v16bf& a, const v16bf& b, const v8f& c) {
    return __builtin_amdgcn_wmma_f32_16x16x32_bf16(false, a, false, b, (short)0, c,
                                                   false, false);
}

// ---------------------------------------------------------------------------
// Kernel 0: f32 -> bf16 pre-conversion of X (=h flattened [4096,1024]) and W.
// ---------------------------------------------------------------------------
__global__ void cvt_kernel(const float* __restrict__ h, const float* __restrict__ w,
                           unsigned short* __restrict__ Xb, unsigned short* __restrict__ Wb) {
    const int stride = gridDim.x * blockDim.x;
    const int tid = blockIdx.x * blockDim.x + threadIdx.x;
    const int NX4 = (MROWS * DMODEL) / 4;
    const int NW4 = (NQKV * DMODEL) / 4;
    for (int i = tid; i < NX4; i += stride) {
        float4 v = reinterpret_cast<const float4*>(h)[i];
        ushort4 o; o.x = f2bf(v.x); o.y = f2bf(v.y); o.z = f2bf(v.z); o.w = f2bf(v.w);
        reinterpret_cast<ushort4*>(Xb)[i] = o;
    }
    for (int i = tid; i < NW4; i += stride) {
        float4 v = reinterpret_cast<const float4*>(w)[i];
        ushort4 o; o.x = f2bf(v.x); o.y = f2bf(v.y); o.z = f2bf(v.z); o.w = f2bf(v.w);
        reinterpret_cast<ushort4*>(Wb)[i] = o;
    }
}

// ---------------------------------------------------------------------------
// Kernel 1: QKV projection. Each wave computes a 16(M) x 64(N) strip of
// heads = X @ W^T with 32 K-steps of v_wmma_f32_16x16x32_bf16 (128 WMMA/wave),
// double-buffered; B rows addressed from ONE base pointer via constant
// immediate offsets (s*16*DMODEL elements) to keep register pressure low.
// Q gets 1/sqrt(d) folded in; V is stored transposed [b,n,d,j] for P@V.
// ---------------------------------------------------------------------------
__global__ void __launch_bounds__(256)
proj_kernel(const unsigned short* __restrict__ Xb,
            const unsigned short* __restrict__ Wb,
            unsigned short* __restrict__ Qb,
            unsigned short* __restrict__ Kb,
            unsigned short* __restrict__ Vt) {
    const int wid  = blockIdx.x * (blockDim.x >> 5) + (threadIdx.x >> 5);
    const int lane = threadIdx.x & 31;
    const int ln = lane & 15, hi = lane >> 4;

    const int nblk   = wid % (NQKV / 64);   // 0..47 -> (t, head)
    const int mstrip = wid / (NQKV / 64);   // 0..255
    const int m0 = mstrip * 16;
    const int c0 = nblk * 64;
    const int t  = nblk >> 4;               // 0=Q 1=K 2=V
    const int n  = nblk & 15;               // head

    // single base pointers; everything else is an immediate offset
    const unsigned short* xrow = Xb + (size_t)(m0 + ln) * DMODEL + hi * 16;
    const unsigned short* wrow = Wb + (size_t)(c0 + ln) * DMODEL + hi * 16;

    v8f zero = {};
    v8f acc[4] = {zero, zero, zero, zero};

    // prologue: k = 0
    BFrag a_cur;
    a_cur.q[0] = *reinterpret_cast<const uint4*>(xrow);
    a_cur.q[1] = *reinterpret_cast<const uint4*>(xrow + 8);
    v16bf b_cur[4];
    #pragma unroll
    for (int s = 0; s < 4; ++s)
        b_cur[s] = *reinterpret_cast<const v16bf*>(wrow + s * 16 * DMODEL);

    #pragma unroll 2
    for (int k = 32; k < DMODEL; k += 32) {
        // prefetch next K-step while WMMAs below execute
        BFrag a_nxt;
        a_nxt.q[0] = *reinterpret_cast<const uint4*>(xrow + k);
        a_nxt.q[1] = *reinterpret_cast<const uint4*>(xrow + k + 8);
        v16bf b_nxt[4];
        #pragma unroll
        for (int s = 0; s < 4; ++s)
            b_nxt[s] = *reinterpret_cast<const v16bf*>(wrow + s * 16 * DMODEL + k);

        #pragma unroll
        for (int s = 0; s < 4; ++s)
            acc[s] = wmma_bf16(a_cur.v, b_cur[s], acc[s]);

        a_cur = a_nxt;
        #pragma unroll
        for (int s = 0; s < 4; ++s) b_cur[s] = b_nxt[s];
    }
    #pragma unroll
    for (int s = 0; s < 4; ++s)
        acc[s] = wmma_bf16(a_cur.v, b_cur[s], acc[s]);

    // scatter results into Q/K (row-major per head) or V (transposed per head)
    #pragma unroll
    for (int s = 0; s < 4; ++s) {
        const int d = s * 16 + ln;
        #pragma unroll
        for (int v = 0; v < 8; ++v) {
            const int m = m0 + v + 8 * hi;
            const int q = m >> 1, bb = m & 1;
            const float val = acc[s][v];
            const size_t head = (size_t)(bb * NHEAD + n);
            if (t == 0)
                Qb[(head * QLEN + q) * DHEAD + d] = f2bf(val * SCALE_Q);
            else if (t == 1)
                Kb[(head * QLEN + q) * DHEAD + d] = f2bf(val);
            else
                Vt[(head * DHEAD + d) * QLEN + q] = f2bf(val);
        }
    }
}

// ---------------------------------------------------------------------------
// Kernel 2: fused attention per (b, n, 16-row i-tile). 256 threads = 8 waves.
//   phase 1: S = Qs * K^T  -> LDS f32 scores[16][2052]     (256 WMMA / WG)
//   phase 2: block softmax over j (2048), f32 probs -> d_out, bf16 -> LDS
//   phase 3: O = P * V, K-dim split across wave halves      (256 WMMA / WG)
// Phases 1 and 3 are double-buffered (prefetch next tile's fragments).
// ---------------------------------------------------------------------------
#define SROW 2052                      // f32 score row stride (bank-padded)
#define PROW 2056                      // bf16 prob row stride (bank-padded)

__global__ void __launch_bounds__(256)
attn_kernel(const unsigned short* __restrict__ Qb,
            const unsigned short* __restrict__ Kb,
            const unsigned short* __restrict__ Vt,
            float* __restrict__ out_vec,
            float* __restrict__ out_prob) {
    extern __shared__ char smem[];
    float*          sc      = reinterpret_cast<float*>(smem);          // 16*2052 f32
    unsigned short* pb      = reinterpret_cast<unsigned short*>(sc + 16 * SROW); // 16*2056 bf16
    float*          red     = reinterpret_cast<float*>(pb + 16 * PROW);// 16*16
    float*          rowstat = red + 256;                               // 16
    float*          scratch = rowstat + 16;                            // 4*256

    const int tid = threadIdx.x;
    const int w = tid >> 5, lane = tid & 31, ln = lane & 15, hi = lane >> 4;
    const int bn = blockIdx.x & 31;                 // 32 (b,n) WGs adjacent per i-tile
    const int b = bn >> 4, n = bn & 15;
    const int i0 = (blockIdx.x >> 5) * 16;

    const size_t head = (size_t)(b * NHEAD + n);
    const unsigned short* Qh = Qb + head * QLEN * DHEAD;
    const unsigned short* Kh = Kb + head * QLEN * DHEAD;
    const unsigned short* Vh = Vt + head * DHEAD * QLEN;

    // ---- Q fragments (rows i0..i0+15, K = d in [0,64)) ----
    const unsigned short* qrow = Qh + (size_t)(i0 + ln) * DHEAD + hi * 16;
    BFrag aq0, aq1;
    aq0.q[0] = *reinterpret_cast<const uint4*>(qrow);
    aq0.q[1] = *reinterpret_cast<const uint4*>(qrow + 8);
    aq1.q[0] = *reinterpret_cast<const uint4*>(qrow + 32);
    aq1.q[1] = *reinterpret_cast<const uint4*>(qrow + 40);

    // ---- phase 1: scores; wave w owns j in [w*256, w*256+256) ----
    {
        const unsigned short* kbase = Kh + (size_t)(w * 256 + ln) * DHEAD + hi * 16;
        v16bf bk0 = *reinterpret_cast<const v16bf*>(kbase);
        v16bf bk1 = *reinterpret_cast<const v16bf*>(kbase + 32);
        #pragma unroll 2
        for (int jt = 0; jt < 15; ++jt) {
            const int koff = (jt + 1) * 16 * DHEAD;   // immediate-friendly offset
            v16bf nk0 = *reinterpret_cast<const v16bf*>(kbase + koff);
            v16bf nk1 = *reinterpret_cast<const v16bf*>(kbase + koff + 32);
            v8f acc = {};
            acc = wmma_bf16(aq0.v, bk0, acc);
            acc = wmma_bf16(aq1.v, bk1, acc);
            const int j0 = w * 256 + jt * 16;
            #pragma unroll
            for (int v = 0; v < 8; ++v)
                sc[(v + 8 * hi) * SROW + j0 + ln] = acc[v];
            bk0 = nk0; bk1 = nk1;
        }
        v8f acc = {};
        acc = wmma_bf16(aq0.v, bk0, acc);
        acc = wmma_bf16(aq1.v, bk1, acc);
        const int j0 = w * 256 + 15 * 16;
        #pragma unroll
        for (int v = 0; v < 8; ++v)
            sc[(v + 8 * hi) * SROW + j0 + ln] = acc[v];
    }
    __syncthreads();

    // ---- phase 2: softmax over j ----
    const int r = tid & 15;            // row within i-tile
    const int ch = tid >> 4;           // 16 column chunks of 128
    const int cbeg = ch * 128;

    float m = -3.402823466e38f;
    for (int c = 0; c < 128; ++c) m = fmaxf(m, sc[r * SROW + cbeg + c]);
    red[r * 16 + ch] = m;
    __syncthreads();
    if (tid < 16) {
        float mm = red[tid * 16];
        for (int k = 1; k < 16; ++k) mm = fmaxf(mm, red[tid * 16 + k]);
        rowstat[tid] = mm;
    }
    __syncthreads();

    const float rm = rowstat[r];
    float sum = 0.f;
    for (int c = 0; c < 128; ++c) {
        float e = __expf(sc[r * SROW + cbeg + c] - rm);
        sc[r * SROW + cbeg + c] = e;
        sum += e;
    }
    red[r * 16 + ch] = sum;
    __syncthreads();
    if (tid < 16) {
        float s = 0.f;
        for (int k = 0; k < 16; ++k) s += red[tid * 16 + k];
        rowstat[tid] = 1.0f / s;
    }
    __syncthreads();

    // normalized probs: f32 -> global [i,j,b,n], bf16 -> LDS for P@V
    {
        const float inv = rowstat[r];
        const size_t ip = (size_t)(i0 + r);
        float* op = out_prob + ((ip * QLEN + cbeg) * BSZ + b) * NHEAD + n;
        for (int c = 0; c < 128; ++c) {
            float p = sc[r * SROW + cbeg + c] * inv;
            op[(size_t)c * (BSZ * NHEAD)] = p;
            pb[r * PROW + cbeg + c] = f2bf(p);
        }
    }
    __syncthreads();

    // ---- phase 3: O = P @ V; wave w -> d-tile (w&3), j-half (w>>2) ----
    const int d0 = (w & 3) * 16;
    const int jh = (w >> 2) * 1024;
    const unsigned short* vrow = Vh + (size_t)(d0 + ln) * QLEN + jh + hi * 16; // Vt[d][j]
    const unsigned short* prow = pb + ln * PROW + jh + hi * 16;
    v8f acc = {};
    {
        BFrag ap;
        ap.q[0] = *reinterpret_cast<const uint4*>(prow);
        ap.q[1] = *reinterpret_cast<const uint4*>(prow + 8);
        v16bf bv = *reinterpret_cast<const v16bf*>(vrow);
        #pragma unroll 2
        for (int s = 0; s < 31; ++s) {
            const int jn = (s + 1) * 32;              // immediate-friendly offset
            BFrag an;
            an.q[0] = *reinterpret_cast<const uint4*>(prow + jn);
            an.q[1] = *reinterpret_cast<const uint4*>(prow + jn + 8);
            v16bf bn2 = *reinterpret_cast<const v16bf*>(vrow + jn);
            acc = wmma_bf16(ap.v, bv, acc);
            ap = an; bv = bn2;
        }
        acc = wmma_bf16(ap.v, bv, acc);
    }
    // reduce the two j-halves, then store attn_vec
    if (w >= 4) {
        #pragma unroll
        for (int v = 0; v < 8; ++v)
            scratch[(w - 4) * 256 + (v + 8 * hi) * 16 + ln] = acc[v];
    }
    __syncthreads();
    if (w < 4) {
        #pragma unroll
        for (int v = 0; v < 8; ++v) {
            const float o = acc[v] + scratch[w * 256 + (v + 8 * hi) * 16 + ln];
            const int irow = i0 + v + 8 * hi;
            out_vec[((size_t)irow * BSZ + b) * (NHEAD * DHEAD) + n * DHEAD + d0 + ln] = o;
        }
    }
}

// ---------------------------------------------------------------------------
extern "C" void kernel_launch(void* const* d_in, const int* in_sizes, int n_in,
                              void* d_out, int out_size, void* d_ws, size_t ws_size,
                              hipStream_t stream) {
    const float* h  = (const float*)d_in[0];
    const float* wq = (const float*)d_in[1];

    float* out_vec  = (float*)d_out;                         // 2048*2*1024
    float* out_prob = out_vec + (size_t)QLEN * BSZ * NHEAD * DHEAD;

    unsigned short* Xb = (unsigned short*)d_ws;              // 4096*1024 bf16
    unsigned short* Wb = Xb + (size_t)MROWS * DMODEL;        // 3072*1024 bf16
    unsigned short* Qb = Wb + (size_t)NQKV * DMODEL;         // [b,n,i,d] bf16
    unsigned short* Kb = Qb + (size_t)BSZ * NHEAD * QLEN * DHEAD;
    unsigned short* Vt = Kb + (size_t)BSZ * NHEAD * QLEN * DHEAD; // [b,n,d,j] bf16

    cvt_kernel<<<2048, 256, 0, stream>>>(h, wq, Xb, Wb);

    // 12288 wave-tasks (256 M-strips x 48 N-blocks), 8 waves / block
    proj_kernel<<<1536, 256, 0, stream>>>(Xb, Wb, Qb, Kb, Vt);

    const size_t shmem = 16 * SROW * 4 + 16 * PROW * 2 + 256 * 4 + 16 * 4 + 4 * 256 * 4;
    // grid: i-tile major, 32 (b,n) minor -> prob-write L2 line locality
    attn_kernel<<<(QLEN / 16) * 32, 256, shmem, stream>>>(Qb, Kb, Vt, out_vec, out_prob);
}
